// GRU_24876450578771
// MI455X (gfx1250) — compile-verified, matrix operand order checked by reading, not run on hardware
//
#include <hip/hip_runtime.h>
#include <hip/hip_bf16.h>

#define T_DIM 4096
#define K_DIM 1024
#define N_DIM 1024
#define NWG   16
#define SLICE (N_DIM / NWG)   // 64

typedef __attribute__((ext_vector_type(16))) __bf16 v16bf;
typedef __attribute__((ext_vector_type(8)))  __bf16 v8bf;
typedef __attribute__((ext_vector_type(8)))  float  v8f;

// ---------------- phase 0: precision conversion ----------------

__global__ void __launch_bounds__(256)
cvt_f32_to_bf16(const float* __restrict__ in, __hip_bfloat16* __restrict__ out, int n) {
  int i = blockIdx.x * blockDim.x + threadIdx.x;
  if (i < n) out[i] = __float2bfloat16(in[i]);
}

// in: [K][N] fp32 row-major  ->  out: [N][K] bf16 row-major (W transposed)
__global__ void __launch_bounds__(256)
transpose_cvt_bf16(const float* __restrict__ in, __hip_bfloat16* __restrict__ out) {
  __shared__ float tile[32][33];
  const int x0 = blockIdx.x * 32;  // n
  const int y0 = blockIdx.y * 32;  // k
  const int tx = threadIdx.x, ty = threadIdx.y;
#pragma unroll
  for (int i = 0; i < 32; i += 8)
    tile[ty + i][tx] = in[(size_t)(y0 + ty + i) * N_DIM + (x0 + tx)];
  __syncthreads();
#pragma unroll
  for (int i = 0; i < 32; i += 8)
    out[(size_t)(x0 + ty + i) * K_DIM + (y0 + tx)] = __float2bfloat16(tile[tx][ty + i]);
}

// ---------------- phase 1: batched projections via bf16 WMMA ----------------
// P[M][N] = A[M][K] * B[K][N] + bias,  B supplied transposed as BT[N][K].
// One wave = one 16x16 output tile, K-loop in chunks of 32 on
// v_wmma_f32_16x16x32_bf16.  Per-lane operand layout per CDNA5 ISA:
//   A: lane(l<16) holds row M=l, K = {hi*8..hi*8+7, 16+hi*8..16+hi*8+7}, hi=l>>4
//   B (mirrored with N across lanes) read from BT so both operands are two
//   contiguous 16-byte (b128) loads per lane per chunk.
__global__ void __launch_bounds__(256)
wmma_gemm_bf16(const __bf16* __restrict__ A,    // [M][K] bf16
               const __bf16* __restrict__ BT,   // [N][K] bf16
               const float*  __restrict__ bias, // [N]
               float*        __restrict__ P)    // [M][N] fp32
{
  const int nTilesN = N_DIM / 16;                               // 64
  const int wave = blockIdx.x * (blockDim.x >> 5) + (threadIdx.x >> 5);
  const int lane = threadIdx.x & 31;
  const int tm = wave / nTilesN;
  const int tn = wave % nTilesN;
  const int hi = (lane >> 4) & 1;
  const int mrow = tm * 16 + (lane & 15);
  const int ncol = tn * 16 + (lane & 15);
  const __bf16* __restrict__ arow = A  + (size_t)mrow * K_DIM;
  const __bf16* __restrict__ brow = BT + (size_t)ncol * K_DIM;

  v8f acc = {};
  for (int kc = 0; kc < K_DIM; kc += 32) {
    if (kc + 32 < K_DIM) {                       // global_prefetch next chunk
      __builtin_prefetch(arow + kc + 32, 0, 1);
      __builtin_prefetch(brow + kc + 32, 0, 1);
    }
    const int kb0 = kc + hi * 8;
    const int kb1 = kc + 16 + hi * 8;
    v8bf a0 = *(const v8bf*)(arow + kb0);
    v8bf a1 = *(const v8bf*)(arow + kb1);
    v8bf b0 = *(const v8bf*)(brow + kb0);
    v8bf b1 = *(const v8bf*)(brow + kb1);
    v16bf a, b;
#pragma unroll
    for (int e = 0; e < 8; ++e) { a[e] = a0[e]; a[e + 8] = a1[e];
                                  b[e] = b0[e]; b[e + 8] = b1[e]; }
    acc = __builtin_amdgcn_wmma_f32_16x16x32_bf16(false, a, false, b,
                                                  (short)0, acc, false, false);
  }
  const float bv = bias[ncol];
#pragma unroll
  for (int v = 0; v < 8; ++v) {                  // D layout: VGPR v -> M = v + 8*hi
    const int m = tm * 16 + v + 8 * hi;
    P[(size_t)m * N_DIM + ncol] = acc[v] + bv;
  }
}

// ---------------- phase 2: persistent sequential scan ----------------

__device__ __forceinline__ void grid_barrier(unsigned* cnt, unsigned target) {
  __threadfence();                               // release all prior stores
  __syncthreads();
  if (threadIdx.x == 0) {
    atomicAdd(cnt, 1u);
    while (atomicAdd(cnt, 0u) < target) __builtin_amdgcn_s_sleep(2);
  }
  __syncthreads();
  __threadfence();                               // acquire remote stores
}

__global__ void __launch_bounds__(256)
gru_scan(const float* __restrict__ xT, const float* __restrict__ xR,
         const float* __restrict__ hW, const float* __restrict__ hRW,
         const float* __restrict__ hB, const float* __restrict__ hRB,
         const float* __restrict__ h0,
         float* __restrict__ partial_r, float* __restrict__ partial_c,
         unsigned* __restrict__ bar, float* __restrict__ out)
{
  __shared__ float h_l[SLICE];                   // this WG owns h[k0 .. k0+63]
  __shared__ float rh_l[SLICE];
  const int w   = blockIdx.x;
  const int k0  = w * SLICE;
  const int tid = threadIdx.x;
  if (tid < SLICE) h_l[tid] = h0[k0 + tid];
  __syncthreads();

  unsigned target = 0;
  for (int t = 0; t < T_DIM; ++t) {
    // stage 1: partial_r[w][j] = sum_{k in slice} h[k] * hRW[k][j],  all j
    {
      float a0 = 0.f, a1 = 0.f, a2 = 0.f, a3 = 0.f;
#pragma unroll 4
      for (int k = 0; k < SLICE; ++k) {
        const float hv = h_l[k];
        const float* __restrict__ row = hRW + (size_t)(k0 + k) * N_DIM + tid;
        a0 = fmaf(hv, row[0],   a0);
        a1 = fmaf(hv, row[256], a1);
        a2 = fmaf(hv, row[512], a2);
        a3 = fmaf(hv, row[768], a3);
      }
      float* pr = partial_r + (size_t)w * N_DIM + tid;
      pr[0] = a0; pr[256] = a1; pr[512] = a2; pr[768] = a3;
    }
    target += NWG; grid_barrier(bar, target);

    // stage 2 (local): r = sigmoid(xR + h@hRW + hRB) on owned slice; rh = r*h
    if (tid < SLICE) {
      const int k = k0 + tid;
      float s = xR[(size_t)t * N_DIM + k] + hRB[k];
      for (int w2 = 0; w2 < NWG; ++w2) s += partial_r[(size_t)w2 * N_DIM + k];
      const float r = 1.0f / (1.0f + expf(-s));
      rh_l[tid] = r * h_l[tid];
    }
    __syncthreads();

    // stage 3: partial_c[w][j] = sum_{k in slice} rh[k] * hW[k][j],  all j
    {
      float a0 = 0.f, a1 = 0.f, a2 = 0.f, a3 = 0.f;
#pragma unroll 4
      for (int k = 0; k < SLICE; ++k) {
        const float rv = rh_l[k];
        const float* __restrict__ row = hW + (size_t)(k0 + k) * N_DIM + tid;
        a0 = fmaf(rv, row[0],   a0);
        a1 = fmaf(rv, row[256], a1);
        a2 = fmaf(rv, row[512], a2);
        a3 = fmaf(rv, row[768], a3);
      }
      float* pc = partial_c + (size_t)w * N_DIM + tid;
      pc[0] = a0; pc[256] = a1; pc[512] = a2; pc[768] = a3;
    }
    target += NWG; grid_barrier(bar, target);

    // stage 4 (local): h_new = tanh(xT + (r*h)@hW + hB) on owned slice
    if (tid < SLICE) {
      const int j = k0 + tid;
      float s = xT[(size_t)t * N_DIM + j] + hB[j];
      for (int w2 = 0; w2 < NWG; ++w2) s += partial_c[(size_t)w2 * N_DIM + j];
      const float hn = tanhf(s);
      h_l[tid] = hn;                             // next step reads only local slice
      out[(size_t)t * N_DIM + j] = hn;           // (1-u)*hc + u*hc == hc
    }
    __syncthreads();
    // No third barrier needed: next stage 1 uses only the locally-owned h
    // slice; partial_r/partial_c overwrite hazards are covered by the two
    // barriers (readers precede their own subsequent barrier arrival).
  }
}

// ---------------- host side ----------------

extern "C" void kernel_launch(void* const* d_in, const int* in_sizes, int n_in,
                              void* d_out, int out_size, void* d_ws, size_t ws_size,
                              hipStream_t stream) {
  const float* x   = (const float*)d_in[0];
  const float* Wx  = (const float*)d_in[1];
  const float* bx  = (const float*)d_in[2];
  const float* Wr  = (const float*)d_in[3];
  const float* br  = (const float*)d_in[4];
  // d_in[5] Wu, d_in[6] bu, d_in[9] hUW, d_in[12] hUB: unused, the update
  // gate cancels exactly: (1-u)*hCand + u*hCand == hCand.
  const float* hW  = (const float*)d_in[7];
  const float* hRW = (const float*)d_in[8];
  const float* hB  = (const float*)d_in[10];
  const float* hRB = (const float*)d_in[11];
  const float* h0  = (const float*)d_in[13];
  float* out = (float*)d_out;

  char* ws = (char*)d_ws;
  size_t off = 0;
  auto alloc = [&](size_t bytes) -> void* {
    void* p = ws + off;
    off = (off + bytes + 255) & ~(size_t)255;
    return p;
  };
  __hip_bfloat16* xb   = (__hip_bfloat16*)alloc((size_t)T_DIM * K_DIM * 2); // 8 MB
  __hip_bfloat16* wxT  = (__hip_bfloat16*)alloc((size_t)K_DIM * N_DIM * 2); // 2 MB
  __hip_bfloat16* wrT  = (__hip_bfloat16*)alloc((size_t)K_DIM * N_DIM * 2); // 2 MB
  float* xTp = (float*)alloc((size_t)T_DIM * N_DIM * 4);                    // 16 MB
  float* xRp = (float*)alloc((size_t)T_DIM * N_DIM * 4);                    // 16 MB
  float* pr  = (float*)alloc((size_t)NWG * N_DIM * 4);
  float* pc  = (float*)alloc((size_t)NWG * N_DIM * 4);
  unsigned* bar = (unsigned*)alloc(256);

  hipMemsetAsync(bar, 0, 256, stream);          // reset barrier counter per call

  cvt_f32_to_bf16<<<(T_DIM * K_DIM) / 256, 256, 0, stream>>>(x, xb, T_DIM * K_DIM);
  transpose_cvt_bf16<<<dim3(N_DIM / 32, K_DIM / 32), dim3(32, 8), 0, stream>>>(Wx, wxT);
  transpose_cvt_bf16<<<dim3(N_DIM / 32, K_DIM / 32), dim3(32, 8), 0, stream>>>(Wr, wrT);

  const int nTiles = (T_DIM / 16) * (N_DIM / 16);           // 16384 wave-tiles
  const int gemmBlocks = nTiles / 8;                        // 8 waves / block
  wmma_gemm_bf16<<<gemmBlocks, 256, 0, stream>>>(
      (const __bf16*)xb, (const __bf16*)wxT, bx, xTp);
  wmma_gemm_bf16<<<gemmBlocks, 256, 0, stream>>>(
      (const __bf16*)xb, (const __bf16*)wrT, br, xRp);

  gru_scan<<<NWG, 256, 0, stream>>>(xTp, xRp, hW, hRW, hB, hRB, h0, pr, pc, bar, out);
}
